// SelfAttention_34067680592337
// MI455X (gfx1250) — compile-verified, matrix operand order checked by reading, not run on hardware
//
#include <hip/hip_runtime.h>

// Problem constants (match reference)
#define BB 8
#define SSQ 2048
#define DDIM 512
#define MTOK (BB * SSQ)        // 16384 total tokens

typedef _Float16 v16h __attribute__((ext_vector_type(16)));
typedef _Float16 v8h  __attribute__((ext_vector_type(8)));
typedef _Float16 v4h  __attribute__((ext_vector_type(4)));
typedef float    v8f  __attribute__((ext_vector_type(8)));

union HFrag { v16h f; v8h h[2]; };

// ---------------------------------------------------------------------------
// Register-blocked wave GEMM: each wave computes a 64x32 C macro-tile as
// 4x2 WMMA 16x16 tiles.  Both operands are row-contiguous along K, so each
// fragment is two global_load_b128 per lane (CDNA5 16-bit fragment layout:
// lane (m=lane%16, half=lane/16) holds K in [half*8,half*8+8) and
// [16+half*8, 16+half*8+8)).  Per 32-deep k-step: 12 b128 loads, 8 WMMAs.
// Register budget: 8 acc (64 VGPR) + 6 frags (48 VGPR) + pipeline buffers
// stays inside the directly-addressable 256-VGPR window (no spills).
// ---------------------------------------------------------------------------
__device__ __forceinline__ void wmma_tile4x2(const _Float16* __restrict__ Abase,
                                             const _Float16* __restrict__ Bbase,
                                             size_t lda, size_t ldb, int kdim,
                                             int half, v8f acc[4][2]) {
    for (int k0 = 0; k0 < kdim; k0 += 32) {
        HFrag a[4], b[2];
        #pragma unroll
        for (int i = 0; i < 4; ++i) {
            const _Float16* p = Abase + (size_t)(i * 16) * lda + k0 + half * 8;
            a[i].h[0] = *(const v8h*)(p);
            a[i].h[1] = *(const v8h*)(p + 16);
        }
        #pragma unroll
        for (int j = 0; j < 2; ++j) {
            const _Float16* p = Bbase + (size_t)(j * 16) * ldb + k0 + half * 8;
            b[j].h[0] = *(const v8h*)(p);
            b[j].h[1] = *(const v8h*)(p + 16);
        }
        #pragma unroll
        for (int i = 0; i < 4; ++i)
            #pragma unroll
            for (int j = 0; j < 2; ++j)
                acc[i][j] = __builtin_amdgcn_wmma_f32_16x16x32_f16(
                    false, a[i].f, false, b[j].f, (short)0, acc[i][j],
                    false, false);
    }
}

// ---------------------------------------------------------------------------
// Prep: f32 -> f16 conversion of x
// ---------------------------------------------------------------------------
__global__ void cvt_f16_kernel(const float* __restrict__ src,
                               _Float16* __restrict__ dst, int n) {
    int i = blockIdx.x * blockDim.x + threadIdx.x;
    if (i < n) dst[i] = (_Float16)src[i];
}

// Prep: W[k][n] (f32) -> Wt[n][k] (f16) for all three weight matrices (z-dim)
__global__ void wtrans_kernel(const float* __restrict__ Wq,
                              const float* __restrict__ Wk,
                              const float* __restrict__ Wv,
                              _Float16* __restrict__ Wqt,
                              _Float16* __restrict__ Wkt,
                              _Float16* __restrict__ Wvt) {
    const float* W = (blockIdx.z == 0) ? Wq : (blockIdx.z == 1) ? Wk : Wv;
    _Float16* Wt   = (blockIdx.z == 0) ? Wqt : (blockIdx.z == 1) ? Wkt : Wvt;
    int n = blockIdx.x * 16 + threadIdx.x;
    int k = blockIdx.y * 16 + threadIdx.y;
    Wt[(size_t)n * DDIM + k] = (_Float16)W[(size_t)k * DDIM + n];
}

// ---------------------------------------------------------------------------
// Projections: Q = x@Wq+bq, K = x@Wk+bk, V = x@Wv+bv (V stored transposed
// per batch: Vt[b][d][s]).
// Block = 256 thr = 8 waves arranged 2(M) x 4(N) -> block tile 128x128.
// grid = (D/128, MTOK/128, 3)
// ---------------------------------------------------------------------------
__global__ void proj_kernel(const _Float16* __restrict__ xh,
                            const _Float16* __restrict__ Wqt,
                            const _Float16* __restrict__ Wkt,
                            const _Float16* __restrict__ Wvt,
                            const float* __restrict__ bq,
                            const float* __restrict__ bk,
                            const float* __restrict__ bv,
                            _Float16* __restrict__ Qh,
                            _Float16* __restrict__ Kh,
                            _Float16* __restrict__ Vth) {
    int which = blockIdx.z;
    const _Float16* Wt  = (which == 0) ? Wqt : (which == 1) ? Wkt : Wvt;
    const float*    bia = (which == 0) ? bq  : (which == 1) ? bk  : bv;

    int lane = threadIdx.x & 31;
    int wave = threadIdx.x >> 5;
    int half = lane >> 4;
    int lm   = lane & 15;
    int waveM = wave >> 2, waveN = wave & 3;

    int mt64 = (blockIdx.y * 2 + waveM) * 64;
    int nt32 = (blockIdx.x * 4 + waveN) * 32;

    const _Float16* Abase = xh + (size_t)(mt64 + lm) * DDIM;
    const _Float16* Bbase = Wt + (size_t)(nt32 + lm) * DDIM;

    v8f acc[4][2] = {};
    wmma_tile4x2(Abase, Bbase, DDIM, DDIM, DDIM, half, acc);

    if (which != 2) {
        _Float16* Out = (which == 0) ? Qh : Kh;
        #pragma unroll
        for (int j = 0; j < 2; ++j) {
            int ncol = nt32 + j * 16 + lm;
            float bval = bia[ncol];
            #pragma unroll
            for (int i = 0; i < 4; ++i) {
                #pragma unroll
                for (int r = 0; r < 8; ++r) {
                    int m = mt64 + i * 16 + half * 8 + r;
                    Out[(size_t)m * DDIM + ncol] =
                        (_Float16)(acc[i][j][r] + bval);
                }
            }
        }
    } else {
        // V transposed per batch: Vth[b*D*S + d*S + s]
        #pragma unroll
        for (int j = 0; j < 2; ++j) {
            int ncol = nt32 + j * 16 + lm;
            float bval = bia[ncol];
            #pragma unroll
            for (int i = 0; i < 4; ++i) {
                #pragma unroll
                for (int r = 0; r < 8; ++r) {
                    int m = mt64 + i * 16 + half * 8 + r;   // global token
                    int b = m >> 11;                        // m / S
                    int s = m & (SSQ - 1);                  // m % S
                    Vth[(size_t)b * DDIM * SSQ + (size_t)ncol * SSQ + s] =
                        (_Float16)(acc[i][j][r] + bval);
                }
            }
        }
    }
}

// ---------------------------------------------------------------------------
// logits = (Q @ K^T) / sqrt(D), stored f32 into the attn output region.
// Block tile 128x128, grid = (S/128, S/128, B)
// ---------------------------------------------------------------------------
__global__ void logits_kernel(const _Float16* __restrict__ Qh,
                              const _Float16* __restrict__ Kh,
                              float* __restrict__ attn) {
    const float scale = 0.044194173824159216f;  // 1/sqrt(512)
    int b    = blockIdx.z;
    int lane = threadIdx.x & 31;
    int wave = threadIdx.x >> 5;
    int half = lane >> 4;
    int lm   = lane & 15;
    int waveM = wave >> 2, waveN = wave & 3;

    int mt64 = (blockIdx.y * 2 + waveM) * 64;   // query offset
    int nt32 = (blockIdx.x * 4 + waveN) * 32;   // key offset

    const _Float16* Abase = Qh + ((size_t)b * SSQ + mt64 + lm) * DDIM;
    const _Float16* Bbase = Kh + ((size_t)b * SSQ + nt32 + lm) * DDIM;

    v8f acc[4][2] = {};
    wmma_tile4x2(Abase, Bbase, DDIM, DDIM, DDIM, half, acc);

    #pragma unroll
    for (int i = 0; i < 4; ++i)
        #pragma unroll
        for (int r = 0; r < 8; ++r) {
            int q = mt64 + i * 16 + half * 8 + r;
            float* rowp = attn + ((size_t)b * SSQ + q) * SSQ + nt32 + lm;
            #pragma unroll
            for (int j = 0; j < 2; ++j)
                rowp[j * 16] = acc[i][j][r] * scale;
        }
}

// ---------------------------------------------------------------------------
// Row softmax over S=2048, single pass through HBM: each of 256 threads keeps
// its 8 row elements (2x float4) in registers; LDS tree reductions for
// max/sum; writes normalized f32 attn + f16 copy Ph exactly once.
// ---------------------------------------------------------------------------
__global__ void softmax_kernel(float* __restrict__ attn,
                               _Float16* __restrict__ Ph) {
    __shared__ float red[256];
    int tid = threadIdx.x;
    size_t base = (size_t)blockIdx.x * SSQ;

    float4* row = (float4*)(attn + base);
    float4 va = row[tid];           // elements [4*tid, 4*tid+4)
    float4 vb = row[tid + 256];     // elements [4*tid+1024, ...)

    float lmax = fmaxf(fmaxf(fmaxf(va.x, va.y), fmaxf(va.z, va.w)),
                       fmaxf(fmaxf(vb.x, vb.y), fmaxf(vb.z, vb.w)));
    red[tid] = lmax; __syncthreads();
    for (int s = 128; s > 0; s >>= 1) {
        if (tid < s) red[tid] = fmaxf(red[tid], red[tid + s]);
        __syncthreads();
    }
    float rmax = red[0]; __syncthreads();

    va.x = __expf(va.x - rmax); va.y = __expf(va.y - rmax);
    va.z = __expf(va.z - rmax); va.w = __expf(va.w - rmax);
    vb.x = __expf(vb.x - rmax); vb.y = __expf(vb.y - rmax);
    vb.z = __expf(vb.z - rmax); vb.w = __expf(vb.w - rmax);

    red[tid] = (va.x + va.y + va.z + va.w) + (vb.x + vb.y + vb.z + vb.w);
    __syncthreads();
    for (int s = 128; s > 0; s >>= 1) {
        if (tid < s) red[tid] += red[tid + s];
        __syncthreads();
    }
    float inv = 1.0f / red[0];

    va.x *= inv; va.y *= inv; va.z *= inv; va.w *= inv;
    vb.x *= inv; vb.y *= inv; vb.z *= inv; vb.w *= inv;
    row[tid]       = va;
    row[tid + 256] = vb;

    v4h* prow = (v4h*)(Ph + base);
    v4h ha = { (_Float16)va.x, (_Float16)va.y, (_Float16)va.z, (_Float16)va.w };
    v4h hb = { (_Float16)vb.x, (_Float16)vb.y, (_Float16)vb.z, (_Float16)vb.w };
    prow[tid]       = ha;
    prow[tid + 256] = hb;
}

// ---------------------------------------------------------------------------
// out = P @ V  (P f16 [S x S], Vt f16 [D x S] per batch -> row-contig in k)
// Block tile 128x128, grid = (D/128, S/128, B)
// ---------------------------------------------------------------------------
__global__ void out_kernel(const _Float16* __restrict__ Ph,
                           const _Float16* __restrict__ Vth,
                           float* __restrict__ out) {
    int b    = threadIdx.y + blockIdx.z;  // placeholder avoided; real below
    b = blockIdx.z;
    int lane = threadIdx.x & 31;
    int wave = threadIdx.x >> 5;
    int half = lane >> 4;
    int lm   = lane & 15;
    int waveM = wave >> 2, waveN = wave & 3;

    int mt64 = (blockIdx.y * 2 + waveM) * 64;   // query offset
    int nt32 = (blockIdx.x * 4 + waveN) * 32;   // feature offset

    const _Float16* Abase = Ph  + ((size_t)b * SSQ + mt64 + lm) * SSQ;
    const _Float16* Bbase = Vth + ((size_t)b * DDIM + nt32 + lm) * SSQ;

    v8f acc[4][2] = {};
    wmma_tile4x2(Abase, Bbase, SSQ, SSQ, SSQ, half, acc);

    #pragma unroll
    for (int i = 0; i < 4; ++i)
        #pragma unroll
        for (int r = 0; r < 8; ++r) {
            int q = mt64 + i * 16 + half * 8 + r;
            float* rowp = out + ((size_t)b * SSQ + q) * DDIM + nt32 + lm;
            #pragma unroll
            for (int j = 0; j < 2; ++j)
                rowp[j * 16] = acc[i][j][r];
        }
}

// ---------------------------------------------------------------------------
// Host launcher
// ---------------------------------------------------------------------------
extern "C" void kernel_launch(void* const* d_in, const int* in_sizes, int n_in,
                              void* d_out, int out_size, void* d_ws, size_t ws_size,
                              hipStream_t stream) {
    const float* x  = (const float*)d_in[0];
    const float* Wq = (const float*)d_in[1];
    const float* bq = (const float*)d_in[2];
    const float* Wk = (const float*)d_in[3];
    const float* bk = (const float*)d_in[4];
    const float* Wv = (const float*)d_in[5];
    const float* bv = (const float*)d_in[6];

    float* out  = (float*)d_out;                       // [B,S,D]
    float* attn = out + (size_t)BB * SSQ * DDIM;       // [B,S,S]

    // Workspace carve-up (all f16)
    char* ws = (char*)d_ws;
    const size_t nx  = (size_t)BB * SSQ * DDIM;        // 8.4M
    const size_t nw  = (size_t)DDIM * DDIM;            // 256K
    const size_t nat = (size_t)BB * SSQ * SSQ;         // 33.5M
    _Float16* xh  = (_Float16*)ws;               ws += nx * 2;
    _Float16* Wqt = (_Float16*)ws;               ws += nw * 2;
    _Float16* Wkt = (_Float16*)ws;               ws += nw * 2;
    _Float16* Wvt = (_Float16*)ws;               ws += nw * 2;
    _Float16* Qh  = (_Float16*)ws;               ws += nx * 2;
    _Float16* Kh  = (_Float16*)ws;               ws += nx * 2;
    _Float16* Vth = (_Float16*)ws;               ws += nx * 2;
    _Float16* Ph  = (_Float16*)ws;               ws += nat * 2;

    // 1. Convert x to f16; transpose+convert weights
    cvt_f16_kernel<<<dim3((unsigned)((nx + 255) / 256)), dim3(256), 0, stream>>>(
        x, xh, (int)nx);
    wtrans_kernel<<<dim3(DDIM / 16, DDIM / 16, 3), dim3(16, 16), 0, stream>>>(
        Wq, Wk, Wv, Wqt, Wkt, Wvt);

    // 2. Q/K/V projections (WMMA, 64x32/wave)
    proj_kernel<<<dim3(DDIM / 128, MTOK / 128, 3), dim3(256), 0, stream>>>(
        xh, Wqt, Wkt, Wvt, bq, bk, bv, Qh, Kh, Vth);

    // 3. logits = QK^T / sqrt(D) (WMMA) -> attn region of d_out
    logits_kernel<<<dim3(SSQ / 128, SSQ / 128, BB), dim3(256), 0, stream>>>(
        Qh, Kh, attn);

    // 4. row softmax; one HBM pass; emits normalized attn + f16 copy Ph
    softmax_kernel<<<dim3(BB * SSQ), dim3(256), 0, stream>>>(attn, Ph);

    // 5. out = P @ V (WMMA, 64x32/wave)
    out_kernel<<<dim3(DDIM / 128, SSQ / 128, BB), dim3(256), 0, stream>>>(
        Ph, Vth, out);
}
